// VectorField_g_44109314130611
// MI455X (gfx1250) — matrix operand order repaired and checked by reading, not compiled, not run on hardware
//
#include <hip/hip_runtime.h>
#include <hip/hip_bf16.h>
#include <math.h>
#include <stdint.h>

// ---------------------------------------------------------------------------
// MI455X / gfx1250. wave32. All GEMMs via v_wmma_f32_16x16x32_f16 (f16 in,
// f32 accumulate). Hot B operands pre-packed into WMMA fragment lane order
// (32 contiguous bytes per lane -> 2 x b128 per fragment); B^T fragments from
// f16 row-major sources are naturally contiguous per lane (2 x b128 direct).
// The [B,N,N] adjacency is built per 16-row tile in LDS and contracted
// immediately (never hits HBM). k_out stages packed W_out via async
// global->LDS DMA (ASYNCcnt path).
// ---------------------------------------------------------------------------

typedef __attribute__((ext_vector_type(16))) _Float16 v16h;
typedef __attribute__((ext_vector_type(8)))  _Float16 v8h;
typedef __attribute__((ext_vector_type(8)))  float    v8f;

#define BATCH 16
#define NN    1024
#define HH    64
#define DD    64
#define MM    128

__device__ __forceinline__ v8f wmma_f16(v16h a, v16h b, v8f c) {
  return __builtin_amdgcn_wmma_f32_16x16x32_f16(
      false, a, false, b, (short)0, c, false, false);
}

// ---- A fragment: 16(M) x 32(K), element (m,k) at src[m*ld + k].
// Lane layout (contiguous K, not ISA table order): lane m%16 holds row m; per lane the 16 halves are
// one contiguous K-run [koff,koff+16), koff=16*(lane/16).
template <typename T>
__device__ __forceinline__ v16h load_a(const T* __restrict__ src, int ld) {
  const int lane = threadIdx.x & 31;
  const int m    = lane & 15;
  const int koff = (lane >> 4) << 4;
  const T* p = src + (size_t)m * ld + koff;
  v16h a;
#pragma unroll
  for (int e = 0; e < 8; ++e) {
    a[e]     = (_Float16)p[e];
    a[8 + e] = (_Float16)p[8 + e];
  }
  return a;
}

// f16 source: two contiguous 16B vector loads (global_load_b128/ds_load_b128).
__device__ __forceinline__ v16h load_a_h(const _Float16* src, int ld) {
  const int lane = threadIdx.x & 31;
  const int m    = lane & 15;
  const int koff = (lane >> 4) << 4;
  const _Float16* p = src + (size_t)m * ld + koff;
  const v8h lo = *(const v8h*)p;
  const v8h hi = *(const v8h*)(p + 8);
  v16h a;
#pragma unroll
  for (int e = 0; e < 8; ++e) { a[e] = lo[e]; a[8 + e] = hi[e]; }
  return a;
}

// ---- B fragment: 32(K) x 16(N), element (k,n) at src[k*ld + n].
// Lanes 0-15: K=0..15, col n=lane; lanes 16-31: K=16..31, col n=lane-16.
template <typename T>
__device__ __forceinline__ v16h load_b(const T* __restrict__ src, int ld) {
  const int lane  = threadIdx.x & 31;
  const int n     = lane & 15;
  const int kbase = (lane >> 4) << 4;
  v16h b;
#pragma unroll
  for (int e = 0; e < 16; ++e)
    b[e] = (_Float16)src[(size_t)(kbase + e) * ld + n];
  return b;
}

// B fragment from transposed (N-major) source: element (k,n) = src[n*ld + k].
template <typename T>
__device__ __forceinline__ v16h load_bt(const T* __restrict__ src, int ld) {
  const int lane  = threadIdx.x & 31;
  const int n     = lane & 15;
  const int kbase = (lane >> 4) << 4;
  v16h b;
#pragma unroll
  for (int e = 0; e < 16; ++e)
    b[e] = (_Float16)src[(size_t)n * ld + kbase + e];
  return b;
}

// f16 transposed source: per lane the 16 halves are CONTIGUOUS -> 2 x b128.
__device__ __forceinline__ v16h load_bt_h(const _Float16* src, int ld) {
  const int lane  = threadIdx.x & 31;
  const int n     = lane & 15;
  const int kbase = (lane >> 4) << 4;
  const _Float16* p = src + (size_t)n * ld + kbase;
  const v8h lo = *(const v8h*)p;
  const v8h hi = *(const v8h*)(p + 8);
  v16h b;
#pragma unroll
  for (int e = 0; e < 8; ++e) { b[e] = lo[e]; b[8 + e] = hi[e]; }
  return b;
}

// Packed-fragment addressing: fragment f, lane l -> 16 contiguous halves at
// pk[(f*32 + l)*16]  (32 B per lane, 1 KB per fragment).
__device__ __forceinline__ v16h load_pk(const _Float16* pk, size_t frag) {
  const int lane = threadIdx.x & 31;
  return *(const v16h*)&pk[(frag * 32 + lane) * 16];
}

// D layout: element e of v8f at (row = e + 8*(lane/16), col = lane%16).
#define D_ROW(e) ((e) + ((threadIdx.x & 31) >> 4 << 3))
#define D_COL    ((threadIdx.x & 31) & 15)

// Async global -> LDS DMA (CDNA5 ISA); tracked by ASYNCcnt.
__device__ __forceinline__ void async_ld_b128(unsigned lds_off, const void* g) {
  asm volatile("global_load_async_to_lds_b128 %0, %1, off"
               :: "v"(lds_off), "v"((unsigned long long)(uintptr_t)g)
               : "memory");
}
__device__ __forceinline__ void wait_async0() {
  asm volatile("s_wait_asynccnt 0x0" ::: "memory");
}

// ---------------------------------------------------------------------------
// Pack kernels: one wave per fragment.
// ---------------------------------------------------------------------------
__global__ void k_pack_wout(const float* __restrict__ W_out,  // [64 x 4096]
                            _Float16* __restrict__ Wout_pk) {
  const int wave = threadIdx.x >> 5;
  const int lane = threadIdx.x & 31;
  const int f = blockIdx.x * (blockDim.x >> 5) + wave;  // 256 ct x 2 kk
  const int ct = f >> 1, kk = f & 1;
  const v16h b = load_b(&W_out[(size_t)(kk * 32) * 4096 + ct * 16], 4096);
  *(v16h*)&Wout_pk[((size_t)f * 32 + lane) * 16] = b;
}

__global__ void k_pack_pool(const float* __restrict__ pool,  // [64 x 8192]
                            _Float16* __restrict__ pool_pk) {
  const int wave = threadIdx.x >> 5;
  const int lane = threadIdx.x & 31;
  const int f = blockIdx.x * (blockDim.x >> 5) + wave;  // 512 ct x 2 kk
  const int ct = f >> 1, kk = f & 1;
  const v16h b = load_b(&pool[(size_t)(kk * 32) * 8192 + ct * 16], 8192);
  *(v16h*)&pool_pk[((size_t)f * 32 + lane) * 16] = b;
}

// Repack h16 (row-major f16) into B-fragment-packed h_pk; each fragment is
// later reused by 64 waves of the K=1024 propagation GEMM.
__global__ void k_repack_h(const _Float16* __restrict__ h16,
                           _Float16* __restrict__ h_pk) {
  const int wave = threadIdx.x >> 5;
  const int lane = threadIdx.x & 31;
  const int f = blockIdx.x * (blockDim.x >> 5) + wave;  // b*128 + kkn*4 + c
  const int b = f >> 7, kkn = (f & 127) >> 2, c = f & 3;
  const v16h bf =
      load_b(&h16[((size_t)b * NN + kkn * 32) * HH + c * 16], HH);
  *(v16h*)&h_pk[((size_t)f * 32 + lane) * 16] = bf;
}

// ---------------------------------------------------------------------------
// Kernel 1 (fused): h = relu(z@W_in+b); q = h@Wq+bq; att = softmax(q Mem^T);
// local = att@Mem. One 16-row tile per wave, 4 waves/block.
// LDS/wave 16 KB: h f16 | q f16 | scores f32 | att f16.
// ---------------------------------------------------------------------------
__global__ void k_fused_in(const float* __restrict__ z,
                           const float* __restrict__ W_in,
                           const float* __restrict__ b_in,
                           const float* __restrict__ Wq_w,
                           const float* __restrict__ Wq_b,
                           const float* __restrict__ Mem,
                           _Float16* __restrict__ h16,
                           _Float16* __restrict__ local16) {
  extern __shared__ char smem[];
  const int wave = threadIdx.x >> 5;
  const int lane = threadIdx.x & 31;
  const int r0   = (blockIdx.x * (blockDim.x >> 5) + wave) * 16;
  _Float16* sh_h   = (_Float16*)(smem + wave * 16384);
  _Float16* sh_q   = (_Float16*)(smem + wave * 16384 + 2048);
  float*    sh_s   = (float*)   (smem + wave * 16384 + 4096);
  _Float16* sh_att = (_Float16*)(smem + wave * 16384 + 12288);

  // h = relu(z @ W_in + b_in)
  v16h az0 = load_a(&z[(size_t)r0 * HH], HH);
  v16h az1 = load_a(&z[(size_t)r0 * HH + 32], HH);
#pragma unroll
  for (int c = 0; c < 4; ++c) {
    v8f acc = {};
    acc = wmma_f16(az0, load_b(&W_in[0 * HH + c * 16], HH), acc);
    acc = wmma_f16(az1, load_b(&W_in[32 * HH + c * 16], HH), acc);
    const int col = c * 16 + D_COL;
    const float bv = b_in[col];
#pragma unroll
    for (int e = 0; e < 8; ++e) {
      float v = acc[e] + bv;
      v = v > 0.f ? v : 0.f;
      sh_h[D_ROW(e) * HH + col] = (_Float16)v;
      h16[(size_t)(r0 + D_ROW(e)) * HH + col] = (_Float16)v;
    }
  }
  // q = h @ Wq + bq
  v16h ah0 = load_a_h(&sh_h[0], HH);
  v16h ah1 = load_a_h(&sh_h[32], HH);
#pragma unroll
  for (int c = 0; c < 4; ++c) {
    v8f acc = {};
    acc = wmma_f16(ah0, load_b(&Wq_w[0 * DD + c * 16], DD), acc);
    acc = wmma_f16(ah1, load_b(&Wq_w[32 * DD + c * 16], DD), acc);
    const int col = c * 16 + D_COL;
    const float bv = Wq_b[col];
#pragma unroll
    for (int e = 0; e < 8; ++e)
      sh_q[D_ROW(e) * DD + col] = (_Float16)(acc[e] + bv);
  }
  // scores = q @ Mem^T (16 x 128)
  v16h aq0 = load_a_h(&sh_q[0], DD);
  v16h aq1 = load_a_h(&sh_q[32], DD);
#pragma unroll
  for (int mb = 0; mb < 8; ++mb) {
    v8f acc = {};
    acc = wmma_f16(aq0, load_bt(&Mem[(mb * 16) * DD], DD), acc);
    acc = wmma_f16(aq1, load_bt(&Mem[(mb * 16) * DD + 32], DD), acc);
#pragma unroll
    for (int e = 0; e < 8; ++e)
      sh_s[D_ROW(e) * MM + mb * 16 + D_COL] = acc[e];
  }
  // row softmax (lanes 0..15 own one row each)
  if (lane < 16) {
    float m = -3.0e38f;
    for (int j = 0; j < MM; ++j) m = fmaxf(m, sh_s[lane * MM + j]);
    float s = 0.f;
    for (int j = 0; j < MM; ++j) s += __expf(sh_s[lane * MM + j] - m);
    const float inv = 1.0f / s;
    for (int j = 0; j < MM; ++j)
      sh_att[lane * MM + j] = (_Float16)(__expf(sh_s[lane * MM + j] - m) * inv);
  }
  // local = att @ Mem (16 x 64)
#pragma unroll
  for (int c = 0; c < 4; ++c) {
    v8f acc = {};
#pragma unroll
    for (int kk = 0; kk < 4; ++kk) {
      v16h a = load_a_h(&sh_att[kk * 32], MM);
      acc = wmma_f16(a, load_b(&Mem[(kk * 32) * DD + c * 16], DD), acc);
    }
    const int col = c * 16 + D_COL;
#pragma unroll
    for (int e = 0; e < 8; ++e)
      local16[(size_t)(r0 + D_ROW(e)) * DD + col] = (_Float16)acc[e];
  }
}

// ---------------------------------------------------------------------------
// Kernel 2: a_g = softmax(relu(E E^T)). 1 wave/block, 16 rows, 64 KB LDS.
// ---------------------------------------------------------------------------
__global__ void k_ag(const float* __restrict__ node_emb,
                     float* __restrict__ ag) {
  extern __shared__ char smem[];
  float* sc = (float*)smem;
  const int lane = threadIdx.x & 31;
  const int r0   = blockIdx.x * 16;
  const v16h a0 = load_a(&node_emb[r0 * DD], DD);
  const v16h a1 = load_a(&node_emb[r0 * DD + 32], DD);
  for (int ct = 0; ct < NN / 16; ++ct) {
    v8f acc = {};
    acc = wmma_f16(a0, load_bt(&node_emb[(ct * 16) * DD], DD), acc);
    acc = wmma_f16(a1, load_bt(&node_emb[(ct * 16) * DD + 32], DD), acc);
    const int col = ct * 16 + D_COL;
#pragma unroll
    for (int e = 0; e < 8; ++e)
      sc[D_ROW(e) * NN + col] = fmaxf(acc[e], 0.f);
  }
  if (lane < 16) {
    float m = -3.0e38f;
    for (int j = 0; j < NN; ++j) m = fmaxf(m, sc[lane * NN + j]);
    float s = 0.f;
    for (int j = 0; j < NN; ++j) s += __expf(sc[lane * NN + j] - m);
    const float inv = 1.0f / s;
    for (int j = 0; j < NN; ++j)
      ag[(size_t)(r0 + lane) * NN + j] = __expf(sc[lane * NN + j] - m) * inv;
  }
}

// ---------------------------------------------------------------------------
// Kernel 3 (fused): scores = relu(local local^T) -> softmax+gate+blend -> A
// row-tile in LDS f16 -> t1 = A @ h (K=1024, A from LDS, B from h_pk).
// B^T fragments of local come straight from row-major f16 (contiguous/lane).
// LDS: 64 KB scores f32 + 32 KB A f16.
// ---------------------------------------------------------------------------
__global__ void k_adj_prop(const _Float16* __restrict__ local16,
                           const _Float16* __restrict__ h_pk,
                           const float* __restrict__ node_emb,
                           const float* __restrict__ ag,
                           _Float16* __restrict__ t116) {
  extern __shared__ char smem[];
  float*    sc = (float*)smem;               // 16 x 1024 f32
  _Float16* sA = (_Float16*)(smem + 65536);  // 16 x 1024 f16
  const int lane = threadIdx.x & 31;
  const int b    = blockIdx.x >> 6;
  const int r0   = (blockIdx.x & 63) * 16;
  const _Float16* Lb = local16 + (size_t)b * NN * DD;

  const v16h a0 = load_a_h(&Lb[r0 * DD], DD);
  const v16h a1 = load_a_h(&Lb[r0 * DD + 32], DD);
  for (int ct = 0; ct < NN / 16; ++ct) {
    v8f acc = {};
    acc = wmma_f16(a0, load_bt_h(&Lb[(ct * 16) * DD], DD), acc);
    acc = wmma_f16(a1, load_bt_h(&Lb[(ct * 16) * DD + 32], DD), acc);
    const int col = ct * 16 + D_COL;
#pragma unroll
    for (int e = 0; e < 8; ++e)
      sc[D_ROW(e) * NN + col] = fmaxf(acc[e], 0.f);
  }
  if (lane < 16) {
    const int r = r0 + lane;
    float g = 0.f;
    for (int d = 0; d < DD; ++d)
      g += (float)Lb[r * DD + d] * node_emb[r * DD + d];
    const float gate = 1.0f / (1.0f + __expf(-g));
    float m = -3.0e38f;
    for (int j = 0; j < NN; ++j) m = fmaxf(m, sc[lane * NN + j]);
    float s = 0.f;
    for (int j = 0; j < NN; ++j) s += __expf(sc[lane * NN + j] - m);
    const float inv = 1.0f / s;
    for (int j = 0; j < NN; ++j) {
      const float al = __expf(sc[lane * NN + j] - m) * inv;
      sA[lane * NN + j] =
          (_Float16)(gate * al + (1.f - gate) * ag[(size_t)r * NN + j]);
    }
  }
  // t1 = A @ h : [16 x 1024] x [1024 x 64]; A from LDS (ds_load_b128 pairs).
  v8f acc[4] = {};
  for (int kk = 0; kk < NN / 32; ++kk) {
    const v16h a = load_a_h(&sA[kk * 32], NN);
#pragma unroll
    for (int c = 0; c < 4; ++c)
      acc[c] = wmma_f16(a, load_pk(h_pk, (size_t)b * 128 + kk * 4 + c), acc[c]);
  }
#pragma unroll
  for (int c = 0; c < 4; ++c) {
    const int col = c * 16 + D_COL;
#pragma unroll
    for (int e = 0; e < 8; ++e)
      t116[((size_t)b * NN + r0 + D_ROW(e)) * HH + col] = (_Float16)acc[c][e];
  }
}

// ---------------------------------------------------------------------------
// Kernel 4: Wn = node_emb @ weights_pool, B from pool_pk (2 x b128/frag),
// output scattered directly into k_h2's B-fragment-packed layout.
// ---------------------------------------------------------------------------
__global__ void k_wpool(const float* __restrict__ node_emb,
                        const _Float16* __restrict__ pool_pk,
                        _Float16* __restrict__ Wn_pk) {
  const int wave = threadIdx.x >> 5;
  const int tile = blockIdx.x * (blockDim.x >> 5) + wave;  // 64 rt x 128 cg
  const int r0 = (tile >> 7) * 16;
  const int cg = tile & 127;
  const int c0 = cg * 64;
  const v16h a0 = load_a(&node_emb[r0 * DD], DD);
  const v16h a1 = load_a(&node_emb[r0 * DD + 32], DD);
  v8f acc[4] = {};
#pragma unroll
  for (int c = 0; c < 4; ++c) {
    const size_t f0 = (size_t)(cg * 4 + c) * 2;
    acc[c] = wmma_f16(a0, load_pk(pool_pk, f0 + 0), acc[c]);
    acc[c] = wmma_f16(a1, load_pk(pool_pk, f0 + 1), acc[c]);
  }
#pragma unroll
  for (int c = 0; c < 4; ++c) {
    const int col = c0 + c * 16 + D_COL;         // flattened (ki, o)
    const int ki = col >> 6, o = col & 63;
    const int kloc = ki & 31;
    const size_t fbase = (size_t)(ki >> 5) * 4 + (o >> 4);  // kk*4 + cf
    const int lane2 = (o & 15) + (kloc & 16);
    const int el = kloc & 15;
#pragma unroll
    for (int e = 0; e < 8; ++e) {
      const int n = r0 + D_ROW(e);
      Wn_pk[(((size_t)n * 16 + fbase) * 32 + lane2) * 16 + el] =
          (_Float16)acc[c][e];
    }
  }
}

// Kernel 5: bias = node_emb @ bias_pool ([1024x64] x [64x64]).
__global__ void k_bias(const float* __restrict__ node_emb,
                       const float* __restrict__ bias_pool,
                       float* __restrict__ biasN) {
  const int wave = threadIdx.x >> 5;
  const int r0 = (blockIdx.x * (blockDim.x >> 5) + wave) * 16;
  const v16h a0 = load_a(&node_emb[r0 * DD], DD);
  const v16h a1 = load_a(&node_emb[r0 * DD + 32], DD);
#pragma unroll
  for (int c = 0; c < 4; ++c) {
    v8f acc = {};
    acc = wmma_f16(a0, load_b(&bias_pool[0 * HH + c * 16], HH), acc);
    acc = wmma_f16(a1, load_b(&bias_pool[32 * HH + c * 16], HH), acc);
    const int col = c * 16 + D_COL;
#pragma unroll
    for (int e = 0; e < 8; ++e)
      biasN[(size_t)(r0 + D_ROW(e)) * HH + col] = acc[e];
  }
}

// ---------------------------------------------------------------------------
// Kernel 6: per-node batched GEMM h2[:,n,:] = [h|t1] (16x128) @ Wn[n] + bias.
// ---------------------------------------------------------------------------
__global__ void k_h2(const _Float16* __restrict__ h16,
                     const _Float16* __restrict__ t116,
                     const _Float16* __restrict__ Wn_pk,
                     const float* __restrict__ biasN,
                     _Float16* __restrict__ h2_16) {
  const int wave = threadIdx.x >> 5;
  const int n = blockIdx.x * (blockDim.x >> 5) + wave;
  v8f acc[4] = {};
#pragma unroll
  for (int kk = 0; kk < 4; ++kk) {
    const _Float16* srcA = (kk < 2) ? &h16[(size_t)n * HH + kk * 32]
                                    : &t116[(size_t)n * HH + (kk - 2) * 32];
    const v16h a = load_a_h(srcA, NN * HH);  // batch stride
#pragma unroll
    for (int c = 0; c < 4; ++c)
      acc[c] = wmma_f16(a, load_pk(Wn_pk, (size_t)n * 16 + kk * 4 + c), acc[c]);
  }
#pragma unroll
  for (int c = 0; c < 4; ++c) {
    const int col = c * 16 + D_COL;
    const float bv = biasN[(size_t)n * HH + col];
#pragma unroll
    for (int e = 0; e < 8; ++e)
      h2_16[((size_t)D_ROW(e) * NN + n) * HH + col] = (_Float16)(acc[c][e] + bv);
  }
}

// ---------------------------------------------------------------------------
// Kernel 7 (dominant, 8.6 GF + 268 MB out): out = tanh(h2 @ W_out + b_out).
// Block = 4 waves sharing one 64-col group; packed W_out (8 KB) staged into
// LDS via async global->LDS DMA, consumed as ds_load_b128 pairs.
// ---------------------------------------------------------------------------
__global__ void k_out(const _Float16* __restrict__ h2_16,
                      const _Float16* __restrict__ Wout_pk,
                      const float* __restrict__ b_out,
                      float* __restrict__ out) {
  extern __shared__ char smem[];
  const int wave = threadIdx.x >> 5;
  const int lane = threadIdx.x & 31;
  const int cg = blockIdx.x & 63;
  const int rt = ((blockIdx.x >> 6) << 2) + wave;
  const int r0 = rt * 16;
  const int c0 = cg * 64;

  // Stage 8 packed fragments (contiguous 8 KB at frag cg*8) into LDS.
  {
    const char* gsrc = (const char*)Wout_pk + (size_t)cg * 8192;
#pragma unroll
    for (int p = 0; p < 4; ++p) {
      const unsigned off = (unsigned)(p * 128 + threadIdx.x) * 16;
      async_ld_b128(off, gsrc + off);
    }
    wait_async0();
  }
  __syncthreads();

  __builtin_prefetch(&h2_16[(size_t)r0 * HH], 0, 3);
  const v16h a0 = load_a_h(&h2_16[(size_t)r0 * HH], HH);
  const v16h a1 = load_a_h(&h2_16[(size_t)r0 * HH + 32], HH);
  const _Float16* sW = (const _Float16*)smem;
#pragma unroll
  for (int c = 0; c < 4; ++c) {
    v8f acc = {};
    const v16h b0 = *(const v16h*)&sW[((c * 2 + 0) * 32 + lane) * 16];
    const v16h b1 = *(const v16h*)&sW[((c * 2 + 1) * 32 + lane) * 16];
    acc = wmma_f16(a0, b0, acc);
    acc = wmma_f16(a1, b1, acc);
    const int col = c0 + c * 16 + D_COL;
    const float bv = b_out[col];
#pragma unroll
    for (int e = 0; e < 8; ++e)
      out[(size_t)(r0 + D_ROW(e)) * 4096 + col] = tanhf(acc[e] + bv);
  }
}

// ---------------------------------------------------------------------------
// Workspace (bytes): 0 h16 2M | 2M local16 2M | 4M t116 2M | 6M h2_16 2M |
// 8M ag(f32) 4M | 12M Wn_pk 16M | 28M biasN 256K | 29M Wout_pk 512K |
// 30M pool_pk 1M | 31M h_pk 2M    (total ~33 MB)
// ---------------------------------------------------------------------------
extern "C" void kernel_launch(void* const* d_in, const int* in_sizes, int n_in,
                              void* d_out, int out_size, void* d_ws,
                              size_t ws_size, hipStream_t stream) {
  (void)in_sizes; (void)n_in; (void)out_size; (void)ws_size;
  const float* z        = (const float*)d_in[0];
  const float* W_in     = (const float*)d_in[1];
  const float* b_in     = (const float*)d_in[2];
  const float* Wq_w     = (const float*)d_in[3];
  const float* Wq_b     = (const float*)d_in[4];
  const float* Mem      = (const float*)d_in[5];
  const float* node_emb = (const float*)d_in[6];
  const float* wpool    = (const float*)d_in[7];
  const float* bpool    = (const float*)d_in[8];
  const float* W_out    = (const float*)d_in[9];
  const float* b_out    = (const float*)d_in[10];
  float* out = (float*)d_out;

  char* ws = (char*)d_ws;
  const size_t MB = 1u << 20;
  _Float16* h16     = (_Float16*)(ws);
  _Float16* local16 = (_Float16*)(ws + 2 * MB);
  _Float16* t116    = (_Float16*)(ws + 4 * MB);
  _Float16* h2_16   = (_Float16*)(ws + 6 * MB);
  float*    ag      = (float*)   (ws + 8 * MB);
  _Float16* Wn_pk   = (_Float16*)(ws + 12 * MB);
  float*    biasN   = (float*)   (ws + 28 * MB);
  _Float16* Wout_pk = (_Float16*)(ws + 29 * MB);
  _Float16* pool_pk = (_Float16*)(ws + 30 * MB);
  _Float16* h_pk    = (_Float16*)(ws + 31 * MB);

  k_pack_wout<<<128, 128, 0, stream>>>(W_out, Wout_pk);
  k_pack_pool<<<256, 128, 0, stream>>>(wpool, pool_pk);
  k_fused_in<<<256, 128, 64 * 1024, stream>>>(z, W_in, b_in, Wq_w, Wq_b, Mem,
                                              h16, local16);
  k_repack_h<<<512, 128, 0, stream>>>(h16, h_pk);
  k_ag<<<64, 32, 64 * 1024, stream>>>(node_emb, ag);
  k_wpool<<<2048, 128, 0, stream>>>(node_emb, pool_pk, Wn_pk);
  k_bias<<<16, 128, 0, stream>>>(node_emb, bpool, biasN);
  k_adj_prop<<<1024, 32, 96 * 1024, stream>>>(local16, h_pk, node_emb, ag,
                                              t116);
  k_h2<<<256, 128, 0, stream>>>(h16, t116, Wn_pk, biasN, h2_16);
  k_out<<<16384, 128, 8192, stream>>>(h2_16, Wout_pk, b_out, out);
}
